// ECGGATModel_89910845374580
// MI455X (gfx1250) — matrix operand order repaired and checked by reading, not compiled, not run on hardware
//
#include <hip/hip_runtime.h>
#include <hip/hip_bf16.h>
#include <cstddef>

typedef _Float16 half_t;
typedef __attribute__((ext_vector_type(16))) _Float16 v16h;
typedef __attribute__((ext_vector_type(8)))  float    v8f;
typedef __attribute__((ext_vector_type(4)))  float    f4;

#define NEG_SLOPE 0.2f

// ---------------------------------------------------------------------------
// Kernel 0: convert W1 (128x256 f32) and W2 (256x64 f32) to f16 and swizzle
// into WMMA B-fragment layout (32x16 tile per fragment):
//   lane n (<16):  column n,        K rows kbase..kbase+15, kbase = kk*32
//   lane n (>=16): column n-16,     kbase = kk*32+16
// 16 halves per lane, K ascending -> one aligned 32B v16h load per fragment.
// W1: 16 n-tiles x 4 k-steps = 64 frags (2048 lane-slots)
// W2:  4 n-tiles x 8 k-steps = 32 frags (1024 lane-slots)
// ---------------------------------------------------------------------------
__global__ void gat_prep_weights(const float* __restrict__ W1,
                                 const float* __restrict__ W2,
                                 half_t* __restrict__ wsW1,
                                 half_t* __restrict__ wsW2) {
    int t = blockIdx.x * blockDim.x + threadIdx.x;
    if (t < 2048) {
        int frag = t >> 5, lane = t & 31;
        int nt = frag >> 2, kk = frag & 3;
        int n = nt * 16 + (lane & 15);
        int kbase = kk * 32 + (lane >> 4) * 16;
        half_t* dst = wsW1 + (size_t)t * 16;
        #pragma unroll
        for (int v = 0; v < 16; ++v)
            dst[v] = (half_t)W1[(kbase + v) * 256 + n];
    } else if (t < 3072) {
        int u = t - 2048;
        int frag = u >> 5, lane = u & 31;
        int nt = frag >> 3, kk = frag & 7;
        int n = nt * 16 + (lane & 15);
        int kbase = kk * 32 + (lane >> 4) * 16;
        half_t* dst = wsW2 + (size_t)u * 16;
        #pragma unroll
        for (int v = 0; v < 16; ++v)
            dst[v] = (half_t)W2[(kbase + v) * 64 + n];
    }
}

// ---------------------------------------------------------------------------
// Fused GAT model: one wave (32 threads) per workgroup, stride loop on batch.
// ---------------------------------------------------------------------------
__global__ __launch_bounds__(32) void gat_fused(
    const float* __restrict__ x,         // [B,12,128]
    const float* __restrict__ att_src1,  // [4,64]
    const float* __restrict__ att_dst1,  // [4,64]
    const float* __restrict__ b1,        // [256]
    const float* __restrict__ att_src2,  // [64]
    const float* __restrict__ att_dst2,  // [64]
    const float* __restrict__ b2,        // [64]
    const float* __restrict__ Wc1,       // [64,32]
    const float* __restrict__ bc1,       // [32]
    const float* __restrict__ Wc2,       // [32]
    const float* __restrict__ bc2,       // [1]
    const half_t* __restrict__ wsW1,     // swizzled f16 W1 fragments
    const half_t* __restrict__ wsW2,     // swizzled f16 W2 fragments
    float* __restrict__ out,             // [B]
    int B) {
    __shared__ float  hbuf[16 * 256];      // h1 (then h2 in first 16*64)
    __shared__ half_t x2h[16 * 256];       // layer-2 input, f16, rows 12..15 = 0
    __shared__ float  alphaBuf[4 * 12 * 12];
    __shared__ float  esed[4 * 16 * 2];    // es1[h*16+j], ed1 at +64
    __shared__ float  esed2[32];           // es2[j], ed2[16+j]
    __shared__ float  alpha2[12 * 12];
    __shared__ float  gbuf[64];
    __shared__ float  red[32];

    const int lane = threadIdx.x;          // 0..31 (wave32)
    const int sub  = lane >> 4;            // 0/1 half-wave
    const int lrow = lane & 15;            // tile row (A) / tile col (B,C)
    const bool rowok = (lrow < 12);

    for (int b = blockIdx.x; b < B; b += gridDim.x) {
        // ---------------- GEMM1: h1 = pad16(x_b) @ W1  (f16 WMMA) ----------
        // A-fragment k order used here (contiguous, same as B): lanes<16 hold K {0..15},
        // lanes>=16 hold K {16..31} of each 32-wide K step.
        v16h afr[4];
        const float* xb = x + ((size_t)b * 12 + lrow) * 128;
        #pragma unroll
        for (int kk = 0; kk < 4; ++kk) {
            int off0 = kk * 32 + sub * 16;
            f4 cc[4] = {{0,0,0,0},{0,0,0,0},{0,0,0,0},{0,0,0,0}};
            if (rowok) {
                cc[0] = *(const f4*)(xb + off0);
                cc[1] = *(const f4*)(xb + off0 + 4);
                cc[2] = *(const f4*)(xb + off0 + 8);
                cc[3] = *(const f4*)(xb + off0 + 12);
            }
            v16h a;
            #pragma unroll
            for (int q = 0; q < 4; ++q) {
                a[q * 4 + 0] = (half_t)cc[q].x;
                a[q * 4 + 1] = (half_t)cc[q].y;
                a[q * 4 + 2] = (half_t)cc[q].z;
                a[q * 4 + 3] = (half_t)cc[q].w;
            }
            afr[kk] = a;
        }
        for (int nt = 0; nt < 16; ++nt) {
            v8f acc = {0, 0, 0, 0, 0, 0, 0, 0};
            #pragma unroll
            for (int kk = 0; kk < 4; ++kk) {
                v16h bfr = *(const v16h*)(wsW1 + ((size_t)((nt * 4 + kk) * 32 + lane) << 4));
                acc = __builtin_amdgcn_wmma_f32_16x16x32_f16(
                    false, afr[kk], false, bfr, (short)0, acc, false, false);
            }
            // C layout: VGPR v -> row v (lanes<16) / v+8 (lanes>=16), col = lrow
            int col = nt * 16 + lrow;
            #pragma unroll
            for (int v = 0; v < 8; ++v)
                hbuf[(v + sub * 8) * 256 + col] = acc[v];
        }
        __syncthreads();

        // ---------------- Attention 1: es/ed per (h, node) -----------------
        if (lane < 12) {
            int j = lane;
            for (int h = 0; h < 4; ++h) {
                float s = 0.f, d0 = 0.f;
                for (int d = 0; d < 64; ++d) {
                    float hv = hbuf[j * 256 + h * 64 + d];
                    s  += hv * att_src1[h * 64 + d];
                    d0 += hv * att_dst1[h * 64 + d];
                }
                esed[h * 16 + j]      = s;
                esed[64 + h * 16 + j] = d0;
            }
        }
        __syncthreads();

        // softmax over sources j for each (h, i): 48 pairs over 32 lanes
        for (int p = lane; p < 48; p += 32) {
            int h = p / 12, i = p % 12;
            float edv = esed[64 + h * 16 + i];
            float z[12], m = -1e30f;
            #pragma unroll
            for (int j = 0; j < 12; ++j) {
                float t = edv + esed[h * 16 + j];
                t = (t > 0.f) ? t : NEG_SLOPE * t;
                z[j] = t;
                m = fmaxf(m, t);
            }
            float ss = 0.f;
            #pragma unroll
            for (int j = 0; j < 12; ++j) { z[j] = __expf(z[j] - m); ss += z[j]; }
            float inv = 1.0f / ss;
            #pragma unroll
            for (int j = 0; j < 12; ++j)
                alphaBuf[(h * 12 + i) * 12 + j] = z[j] * inv;
        }
        __syncthreads();

        // ------------- aggregate + b1 + ELU -> x2 (f16, padded) -------------
        for (int c = lane; c < 256; c += 32) {
            int h = c >> 6;
            float bb = b1[c];
            for (int i = 0; i < 12; ++i) {
                const float* al = &alphaBuf[(h * 12 + i) * 12];
                float s = 0.f;
                #pragma unroll
                for (int j = 0; j < 12; ++j) s += al[j] * hbuf[j * 256 + c];
                float v = s + bb;
                v = (v > 0.f) ? v : (__expf(v) - 1.0f);   // ELU
                x2h[i * 256 + c] = (half_t)v;
            }
            #pragma unroll
            for (int i = 12; i < 16; ++i) x2h[i * 256 + c] = (half_t)0.f;
        }
        __syncthreads();

        // ---------------- GEMM2: h2 = x2 @ W2 (f16 WMMA) --------------------
        for (int nt = 0; nt < 4; ++nt) {
            v8f acc = {0, 0, 0, 0, 0, 0, 0, 0};
            #pragma unroll
            for (int kk = 0; kk < 8; ++kk) {
                int off0 = kk * 32 + sub * 16;
                const half_t* xr = &x2h[lrow * 256];
                v16h a;
                #pragma unroll
                for (int v = 0; v < 8; ++v) a[v]     = xr[off0 + v];
                #pragma unroll
                for (int v = 0; v < 8; ++v) a[8 + v] = xr[off0 + 8 + v];
                v16h bfr = *(const v16h*)(wsW2 + ((size_t)((nt * 8 + kk) * 32 + lane) << 4));
                acc = __builtin_amdgcn_wmma_f32_16x16x32_f16(
                    false, a, false, bfr, (short)0, acc, false, false);
            }
            int col = nt * 16 + lrow;
            #pragma unroll
            for (int v = 0; v < 8; ++v)
                hbuf[(v + sub * 8) * 64 + col] = acc[v];   // h2 in hbuf[0..1023]
        }
        __syncthreads();

        // ---------------- Attention 2 (single head) -------------------------
        if (lane < 12) {
            int j = lane;
            float s = 0.f, d0 = 0.f;
            for (int d = 0; d < 64; ++d) {
                float hv = hbuf[j * 64 + d];
                s  += hv * att_src2[d];
                d0 += hv * att_dst2[d];
            }
            esed2[j]      = s;
            esed2[16 + j] = d0;
        }
        __syncthreads();
        if (lane < 12) {
            int i = lane;
            float edv = esed2[16 + i];
            float z[12], m = -1e30f;
            #pragma unroll
            for (int j = 0; j < 12; ++j) {
                float t = edv + esed2[j];
                t = (t > 0.f) ? t : NEG_SLOPE * t;
                z[j] = t;
                m = fmaxf(m, t);
            }
            float ss = 0.f;
            #pragma unroll
            for (int j = 0; j < 12; ++j) { z[j] = __expf(z[j] - m); ss += z[j]; }
            float inv = 1.0f / ss;
            #pragma unroll
            for (int j = 0; j < 12; ++j) alpha2[i * 12 + j] = z[j] * inv;
        }
        __syncthreads();

        // -------- aggregate2 + b2 + ELU + global mean pool over nodes -------
        for (int c = lane; c < 64; c += 32) {
            float bb = b2[c];
            float accp = 0.f;
            for (int i = 0; i < 12; ++i) {
                float s = 0.f;
                #pragma unroll
                for (int j = 0; j < 12; ++j) s += alpha2[i * 12 + j] * hbuf[j * 64 + c];
                float v = s + bb;
                v = (v > 0.f) ? v : (__expf(v) - 1.0f);   // ELU
                accp += v;
            }
            gbuf[c] = accp * (1.0f / 12.0f);
        }
        __syncthreads();

        // ---------------- classifier: relu(g@Wc1+bc1) @ Wc2 + bc2 ----------
        {
            float s = bc1[lane];
            for (int d = 0; d < 64; ++d) s += gbuf[d] * Wc1[d * 32 + lane];
            s = fmaxf(s, 0.f);
            red[lane] = s * Wc2[lane];
        }
        __syncthreads();
        if (lane == 0) {
            float y = bc2[0];
            #pragma unroll
            for (int k = 0; k < 32; ++k) y += red[k];
            out[b] = y;
        }
        __syncthreads();   // protect LDS reuse in next batch iteration
    }
}

extern "C" void kernel_launch(void* const* d_in, const int* in_sizes, int n_in,
                              void* d_out, int out_size, void* d_ws, size_t ws_size,
                              hipStream_t stream) {
    const float* x        = (const float*)d_in[0];   // [B,12,128]
    const float* W1       = (const float*)d_in[1];   // [128,256]
    const float* att_src1 = (const float*)d_in[2];   // [4,64]
    const float* att_dst1 = (const float*)d_in[3];   // [4,64]
    const float* b1       = (const float*)d_in[4];   // [256]
    const float* W2       = (const float*)d_in[5];   // [256,64]
    const float* att_src2 = (const float*)d_in[6];   // [1,64]
    const float* att_dst2 = (const float*)d_in[7];   // [1,64]
    const float* b2       = (const float*)d_in[8];   // [64]
    const float* Wc1      = (const float*)d_in[9];   // [64,32]
    const float* bc1      = (const float*)d_in[10];  // [32]
    const float* Wc2      = (const float*)d_in[11];  // [32,1]
    const float* bc2      = (const float*)d_in[12];  // [1]

    const int B = in_sizes[0] / (12 * 128);

    half_t* wsW1 = (half_t*)d_ws;                    // 2048 slots * 16 halves
    half_t* wsW2 = wsW1 + 2048 * 16;                 // 1024 slots * 16 halves

    gat_prep_weights<<<12, 256, 0, stream>>>(W1, W2, wsW1, wsW2);

    int grid = 4096;
    if (grid > B) grid = B;
    gat_fused<<<grid, 32, 0, stream>>>(x, att_src1, att_dst1, b1,
                                       att_src2, att_dst2, b2,
                                       Wc1, bc1, Wc2, bc2,
                                       wsW1, wsW2, (float*)d_out, B);
}
